// KGATAttention_41686952575547
// MI455X (gfx1250) — compile-verified, matrix operand order checked by reading, not run on hardware
//
#include <hip/hip_runtime.h>
#include <hip/hip_bf16.h>

// CDNA5 / gfx1250: wave32, WMMA 16x16x32 bf16 -> f32
typedef __attribute__((ext_vector_type(16))) __bf16 v16bf;
typedef __attribute__((ext_vector_type(8)))  float  v8f;

#define D    64
#define WSZ  32

__device__ __forceinline__ float leaky02(float x) { return x > 0.f ? x : 0.2f * x; }

// monotone float<->uint encoding for atomicMax over signed floats
__device__ __forceinline__ unsigned encf(float f) {
    unsigned u = __float_as_uint(f);
    return (u & 0x80000000u) ? ~u : (u | 0x80000000u);
}
__device__ __forceinline__ float decf(unsigned e) {
    unsigned u = (e & 0x80000000u) ? (e & 0x7fffffffu) : ~e;
    return __uint_as_float(u);
}

// ---------------------------------------------------------------- init ws ---
__global__ void k_init(float* denom, float* hneigh, unsigned* maxenc, int N) {
    long long i = (long long)blockIdx.x * blockDim.x + threadIdx.x;
    long long stride = (long long)gridDim.x * blockDim.x;
    long long total = (long long)N * D;
    for (long long j = i; j < total; j += stride) hneigh[j] = 0.f;
    for (long long j = i; j < N; j += stride) denom[j] = 0.f;
    if (i == 0) maxenc[0] = encf(-3.4e38f);
}

// ---------------------------------------------- node transform (WMMA x2) ---
// h_att = feats@W_att + b_att ; s_src = h_att@a[:D] ; s_dst = h_att@a[D:]
// h_msg = feats@W1 + b1  (stored f32 [N,64])
__global__ __launch_bounds__(256) void k_node_transform(
    const float* __restrict__ user_embed, const float* __restrict__ ent_embed,
    const float* __restrict__ Watt, const float* __restrict__ Watt_b,
    const float* __restrict__ W1,   const float* __restrict__ W1_b,
    const float* __restrict__ avec,
    float* __restrict__ s_src, float* __restrict__ s_dst,
    float* __restrict__ hmsg, int n_users, int n_nodes)
{
    __shared__ __bf16 WtA[D * D];  // transposed: WtA[n*64+k] = Watt[k][n]
    __shared__ __bf16 Wt1[D * D];
    for (int i = threadIdx.x; i < D * D; i += blockDim.x) {
        int k = i >> 6, n = i & 63;
        WtA[n * D + k] = (__bf16)Watt[i];
        Wt1[n * D + k] = (__bf16)W1[i];
    }
    __syncthreads();

    const int wid  = threadIdx.x >> 5;
    const int lane = threadIdx.x & 31;
    const int tile = blockIdx.x * 8 + wid;
    const int ntiles = n_nodes >> 4;   // N is a multiple of 16
    if (tile >= ntiles) return;

    const int base = tile << 4;
    const int g    = lane >> 4;        // lane half -> K-run select
    const int mlo  = lane & 15;        // A row / C column (N) index

    const int node_a = base + mlo;
    const float* rowA = (node_a < n_users)
        ? user_embed + (size_t)node_a * D
        : ent_embed + (size_t)(node_a - n_users) * D;

    v8f zero = {};
    v8f accA[4] = {zero, zero, zero, zero};
    v8f accM[4] = {zero, zero, zero, zero};

#pragma unroll
    for (int kb = 0; kb < 2; ++kb) {
        const int k0 = kb * 32 + g * 8;        // K runs for this lane half
        const int k1 = kb * 32 + 16 + g * 8;
        float tf[16];
        *(float4*)(tf + 0)  = *(const float4*)(rowA + k0);
        *(float4*)(tf + 4)  = *(const float4*)(rowA + k0 + 4);
        *(float4*)(tf + 8)  = *(const float4*)(rowA + k1);
        *(float4*)(tf + 12) = *(const float4*)(rowA + k1 + 4);
        v16bf afrag;
#pragma unroll
        for (int j = 0; j < 16; ++j) afrag[j] = (__bf16)tf[j];

#pragma unroll
        for (int nt = 0; nt < 4; ++nt) {
            const int n = nt * 16 + mlo;
            v16bf bA, bM;
#pragma unroll
            for (int j = 0; j < 8; ++j) {
                bA[j]     = WtA[n * D + k0 + j];
                bA[8 + j] = WtA[n * D + k1 + j];
                bM[j]     = Wt1[n * D + k0 + j];
                bM[8 + j] = Wt1[n * D + k1 + j];
            }
            accA[nt] = __builtin_amdgcn_wmma_f32_16x16x32_bf16(
                false, afrag, false, bA, (short)0, accA[nt], false, false);
            accM[nt] = __builtin_amdgcn_wmma_f32_16x16x32_bf16(
                false, afrag, false, bM, (short)0, accM[nt], false, false);
        }
    }

    // epilogue: bias, a-projections, store h_msg
    float psS[8], psD[8];
#pragma unroll
    for (int i = 0; i < 8; ++i) { psS[i] = 0.f; psD[i] = 0.f; }

#pragma unroll
    for (int nt = 0; nt < 4; ++nt) {
        const int n = nt * 16 + mlo;
        const float bA = Watt_b[n], b1 = W1_b[n];
        const float as = avec[n],   ad = avec[D + n];
#pragma unroll
        for (int i = 0; i < 8; ++i) {
            const int m = i + 8 * g;                 // C/D layout row
            float hA = accA[nt][i] + bA;
            float hM = accM[nt][i] + b1;
            psS[i] += hA * as;
            psD[i] += hA * ad;
            hmsg[(size_t)(base + m) * D + n] = hM;
        }
    }
    // reduce over the 16 lanes of each half (n dimension)
#pragma unroll
    for (int msk = 1; msk < 16; msk <<= 1) {
#pragma unroll
        for (int i = 0; i < 8; ++i) {
            psS[i] += __shfl_xor(psS[i], msk, WSZ);
            psD[i] += __shfl_xor(psD[i], msk, WSZ);
        }
    }
    if (mlo == 0) {
#pragma unroll
        for (int i = 0; i < 8; ++i) {
            const int m = i + 8 * g;
            s_src[base + m] = psS[i];
            s_dst[base + m] = psD[i];
        }
    }
}

// -------------------------------------------------------- edge: global max --
__global__ __launch_bounds__(256) void k_edge_max(
    const int* __restrict__ idx, const float* __restrict__ s_src,
    const float* __restrict__ s_dst, unsigned* maxenc, int E)
{
    __shared__ unsigned red[256];
    long long i0 = (long long)blockIdx.x * blockDim.x + threadIdx.x;
    long long stride = (long long)gridDim.x * blockDim.x;
    float m = -3.4e38f;
    for (long long e = i0; e < E; e += stride) {
        float v = leaky02(s_src[idx[e]] + s_dst[idx[(long long)E + e]]);
        m = fmaxf(m, v);
    }
    red[threadIdx.x] = encf(m);
    __syncthreads();
    for (int s = 128; s > 0; s >>= 1) {
        if ((int)threadIdx.x < s) {
            unsigned a = red[threadIdx.x], b = red[threadIdx.x + s];
            red[threadIdx.x] = a > b ? a : b;
        }
        __syncthreads();
    }
    if (threadIdx.x == 0) atomicMax(maxenc, red[0]);
}

// ---------------------------------------------------- edge: softmax denom ---
__global__ __launch_bounds__(256) void k_edge_denom(
    const int* __restrict__ idx, const float* __restrict__ s_src,
    const float* __restrict__ s_dst, const unsigned* __restrict__ maxenc,
    float* denom, int E)
{
    long long e = (long long)blockIdx.x * blockDim.x + threadIdx.x;
    if (e >= E) return;
    float mx = decf(*maxenc);
    int s = idx[e], d = idx[(long long)E + e];
    float ex = __expf(leaky02(s_src[s] + s_dst[d]) - mx);
    __hip_atomic_fetch_add(&denom[d], ex, __ATOMIC_RELAXED, __HIP_MEMORY_SCOPE_AGENT);
}

// ------------------------------------------- edge: weighted scatter-add ------
// one wave per edge; lanes cover 2 dims each; h_msg/h_neigh are L2-resident
__global__ __launch_bounds__(256) void k_edge_aggregate(
    const int* __restrict__ idx, const float* __restrict__ s_src,
    const float* __restrict__ s_dst, const unsigned* __restrict__ maxenc,
    const float* __restrict__ denom, const float* __restrict__ hmsg,
    float* hneigh, int E)
{
    const int lane = threadIdx.x & 31;
    long long w  = ((long long)blockIdx.x * blockDim.x + threadIdx.x) >> 5;
    long long nw = ((long long)gridDim.x * blockDim.x) >> 5;
    const float mx = decf(*maxenc);
    for (long long e = w; e < E; e += nw) {
        int s = idx[e], d = idx[(long long)E + e];
        float alpha = __expf(leaky02(s_src[s] + s_dst[d]) - mx) / (denom[d] + 1e-9f);
        const float* hm = hmsg + (size_t)s * D;
        float* hn = hneigh + (size_t)d * D;
        float2 v = *(const float2*)(hm + lane * 2);
        __hip_atomic_fetch_add(&hn[lane * 2],     v.x * alpha, __ATOMIC_RELAXED, __HIP_MEMORY_SCOPE_AGENT);
        __hip_atomic_fetch_add(&hn[lane * 2 + 1], v.y * alpha, __ATOMIC_RELAXED, __HIP_MEMORY_SCOPE_AGENT);
    }
}

// --------------------------- h_out for the 2*B gathered rows only (WMMA) ----
__global__ __launch_bounds__(256) void k_hout(
    const int* __restrict__ user_ids, const int* __restrict__ item_ids,
    const float* __restrict__ user_embed, const float* __restrict__ ent_embed,
    const float* __restrict__ W2, const float* __restrict__ W2_b,
    const float* __restrict__ hneigh, float* __restrict__ hout,
    int n_users, int B)
{
    __shared__ __bf16 Wt2[D * D];
    for (int i = threadIdx.x; i < D * D; i += blockDim.x) {
        int k = i >> 6, n = i & 63;
        Wt2[n * D + k] = (__bf16)W2[i];
    }
    __syncthreads();

    const int wid  = threadIdx.x >> 5;
    const int lane = threadIdx.x & 31;
    const int tile = blockIdx.x * 8 + wid;
    const int ntiles = (2 * B) >> 4;
    if (tile >= ntiles) return;

    const int base = tile << 4;
    const int g    = lane >> 4;
    const int mlo  = lane & 15;

    // A-fragment row for this lane
    const int ra = base + mlo;
    const int node_a = (ra < B) ? user_ids[ra] : n_users + item_ids[ra - B];
    const float* fA = (node_a < n_users)
        ? user_embed + (size_t)node_a * D
        : ent_embed + (size_t)(node_a - n_users) * D;
    const float* hA = hneigh + (size_t)node_a * D;

    v8f zero = {};
    v8f acc[4] = {zero, zero, zero, zero};

#pragma unroll
    for (int kb = 0; kb < 2; ++kb) {
        const int k0 = kb * 32 + g * 8;
        const int k1 = kb * 32 + 16 + g * 8;
        float tf[16], th[16];
        *(float4*)(tf + 0)  = *(const float4*)(fA + k0);
        *(float4*)(tf + 4)  = *(const float4*)(fA + k0 + 4);
        *(float4*)(tf + 8)  = *(const float4*)(fA + k1);
        *(float4*)(tf + 12) = *(const float4*)(fA + k1 + 4);
        *(float4*)(th + 0)  = *(const float4*)(hA + k0);
        *(float4*)(th + 4)  = *(const float4*)(hA + k0 + 4);
        *(float4*)(th + 8)  = *(const float4*)(hA + k1);
        *(float4*)(th + 12) = *(const float4*)(hA + k1 + 4);
        v16bf afrag;
#pragma unroll
        for (int j = 0; j < 16; ++j) afrag[j] = (__bf16)(tf[j] * th[j]);  // prod_h

#pragma unroll
        for (int nt = 0; nt < 4; ++nt) {
            const int n = nt * 16 + mlo;
            v16bf bf;
#pragma unroll
            for (int j = 0; j < 8; ++j) {
                bf[j]     = Wt2[n * D + k0 + j];
                bf[8 + j] = Wt2[n * D + k1 + j];
            }
            acc[nt] = __builtin_amdgcn_wmma_f32_16x16x32_bf16(
                false, afrag, false, bf, (short)0, acc[nt], false, false);
        }
    }

    // epilogue rows in C/D layout: m = i + 8*g
    const float* fC[8]; const float* hC[8];
#pragma unroll
    for (int i = 0; i < 8; ++i) {
        const int r = base + i + 8 * g;
        const int nd = (r < B) ? user_ids[r] : n_users + item_ids[r - B];
        fC[i] = (nd < n_users) ? user_embed + (size_t)nd * D
                               : ent_embed + (size_t)(nd - n_users) * D;
        hC[i] = hneigh + (size_t)nd * D;
    }

    float tv[4][8];
    float ssq[8];
#pragma unroll
    for (int i = 0; i < 8; ++i) ssq[i] = 0.f;

#pragma unroll
    for (int nt = 0; nt < 4; ++nt) {
        const int n = nt * 16 + mlo;
        const float bb = W2_b[n];
#pragma unroll
        for (int i = 0; i < 8; ++i) {
            float f = fC[i][n];
            float hn = hC[i][n];
            float t = acc[nt][i] + bb + f + hn;   // (prod@W2 + b) + sum_h
            t = leaky02(t);
            tv[nt][i] = t;
            ssq[i] += t * t;
        }
    }
#pragma unroll
    for (int msk = 1; msk < 16; msk <<= 1)
#pragma unroll
        for (int i = 0; i < 8; ++i) ssq[i] += __shfl_xor(ssq[i], msk, WSZ);

    float sc[8];
#pragma unroll
    for (int i = 0; i < 8; ++i) sc[i] = 1.f / fmaxf(sqrtf(ssq[i]), 1e-12f);

#pragma unroll
    for (int nt = 0; nt < 4; ++nt) {
        const int n = nt * 16 + mlo;
#pragma unroll
        for (int i = 0; i < 8; ++i)
            hout[(size_t)(base + i + 8 * g) * D + n] = tv[nt][i] * sc[i];
    }
}

// -------------------------------------------------- final pairwise scores ---
__global__ __launch_bounds__(256) void k_score(
    const int* __restrict__ user_ids, const int* __restrict__ item_ids,
    const float* __restrict__ user_embed, const float* __restrict__ ent_embed,
    const float* __restrict__ hout, float* __restrict__ out, int B)
{
    const int lane = threadIdx.x & 31;
    const int w = (blockIdx.x * blockDim.x + threadIdx.x) >> 5;
    if (w >= B) return;
    const float* fu = user_embed + (size_t)user_ids[w] * D;
    const float* fv = ent_embed + (size_t)item_ids[w] * D;
    const float* hu = hout + (size_t)w * D;
    const float* hv = hout + (size_t)(B + w) * D;
    float s = fu[lane] * fv[lane] + fu[lane + 32] * fv[lane + 32]
            + hu[lane] * hv[lane] + hu[lane + 32] * hv[lane + 32];
#pragma unroll
    for (int msk = 16; msk > 0; msk >>= 1) s += __shfl_xor(s, msk, WSZ);
    if (lane == 0) out[w] = s;
}

// ---------------------------------------------------------------------------
extern "C" void kernel_launch(void* const* d_in, const int* in_sizes, int n_in,
                              void* d_out, int out_size, void* d_ws, size_t ws_size,
                              hipStream_t stream) {
    (void)n_in; (void)out_size; (void)ws_size;
    const int*   indices    = (const int*)d_in[0];
    const int*   user_ids   = (const int*)d_in[1];
    const int*   item_ids   = (const int*)d_in[2];
    /* d_in[3] = num_nodes scalar (derived from sizes instead) */
    const float* user_embed = (const float*)d_in[4];
    const float* ent_embed  = (const float*)d_in[5];
    const float* Watt       = (const float*)d_in[6];
    const float* Watt_b     = (const float*)d_in[7];
    const float* W1         = (const float*)d_in[8];
    const float* W1_b       = (const float*)d_in[9];
    const float* W2         = (const float*)d_in[10];
    const float* W2_b       = (const float*)d_in[11];
    const float* avec       = (const float*)d_in[12];
    float* out = (float*)d_out;

    const int E = in_sizes[0] / 2;
    const int B = in_sizes[1];
    const int n_users = in_sizes[4] / D;
    const int n_ent   = in_sizes[5] / D;
    const int N = n_users + n_ent;   // 200,000 (multiple of 16)

    char* ws = (char*)d_ws;
    size_t off = 0;
    auto walloc = [&](size_t bytes) -> void* {
        void* p = ws + off;
        off += (bytes + 255) & ~(size_t)255;
        return p;
    };
    float*    s_src  = (float*)walloc((size_t)N * 4);
    float*    s_dst  = (float*)walloc((size_t)N * 4);
    float*    hmsg   = (float*)walloc((size_t)N * D * 4);
    float*    denom  = (float*)walloc((size_t)N * 4);
    float*    hneigh = (float*)walloc((size_t)N * D * 4);
    unsigned* maxenc = (unsigned*)walloc(256);
    float*    hout   = (float*)walloc((size_t)2 * B * D * 4);

    k_init<<<2048, 256, 0, stream>>>(denom, hneigh, maxenc, N);

    const int ntiles = N / 16;
    k_node_transform<<<(ntiles + 7) / 8, 256, 0, stream>>>(
        user_embed, ent_embed, Watt, Watt_b, W1, W1_b, avec,
        s_src, s_dst, hmsg, n_users, N);

    k_edge_max<<<2048, 256, 0, stream>>>(indices, s_src, s_dst, maxenc, E);
    k_edge_denom<<<(E + 255) / 256, 256, 0, stream>>>(
        indices, s_src, s_dst, maxenc, denom, E);
    k_edge_aggregate<<<4096, 256, 0, stream>>>(
        indices, s_src, s_dst, maxenc, denom, hmsg, hneigh, E);

    const int gtiles = (2 * B) / 16;
    k_hout<<<(gtiles + 7) / 8, 256, 0, stream>>>(
        user_ids, item_ids, user_embed, ent_embed, W2, W2_b,
        hneigh, hout, n_users, B);

    k_score<<<(B + 7) / 8, 256, 0, stream>>>(
        user_ids, item_ids, user_embed, ent_embed, hout, out, B);
}
